// MPNNModel_31267361914920
// MI455X (gfx1250) — matrix operand order repaired and checked by reading, not compiled, not run on hardware
//
#include <hip/hip_runtime.h>
#include <hip/hip_bf16.h>

// ---------------------------------------------------------------------------
// MPNN on MI455X (gfx1250): bf16 WMMA GEMMs, TDM weight staging into LDS,
// LDS-transposed coalesced epilogues, fused BatchNorm, atomic scatter.
// wave32: each wave owns one 16-row tile, 4 WMMA N-tiles (64 features).
// ---------------------------------------------------------------------------

typedef __attribute__((ext_vector_type(16))) __bf16 bf16x16;
typedef __attribute__((ext_vector_type(8)))  float  f32x8;
typedef __attribute__((ext_vector_type(4)))  unsigned int u32x4;
typedef __attribute__((ext_vector_type(8)))  int    i32x8;
typedef __attribute__((ext_vector_type(4)))  int    i32x4;

#define DEV_INLINE __device__ __forceinline__

#if defined(__has_builtin)
#if __has_builtin(__builtin_amdgcn_tensor_load_to_lds)
#define HAVE_TDM 1
#endif
#if __has_builtin(__builtin_amdgcn_s_wait_tensorcnt)
#define HAVE_TENSORCNT 1
#endif
#endif

DEV_INLINE f32x8 wmma_bf16(bf16x16 a, bf16x16 b, f32x8 c) {
  return __builtin_amdgcn_wmma_f32_16x16x32_bf16(
      /*neg_a=*/false, a, /*neg_b=*/false, b,
      /*c_mod=*/(short)0, c, /*reuse_a=*/false, /*reuse_b=*/false);
}

// A fragment from a row-major bf16 row (32-wide K chunk), ISA 16-bit A layout:
//   lanes 0-15  (hi=0): elems 0-7 -> K=0..7,  elems 8-15 -> K=16..23
//   lanes 16-31 (hi=1): elems 0-7 -> K=8..15, elems 8-15 -> K=24..31
DEV_INLINE bf16x16 loadA_bf16row(const __bf16* row, int hi) {
  union { bf16x16 v; uint4 q[2]; } u;
  const uint4* p = (const uint4*)row;
  u.q[0] = p[hi];
  u.q[1] = p[2 + hi];
  return u.v;
}

// A fragment from a row-major fp32 row, converting to bf16; zero-pad K >= kvalid.
DEV_INLINE bf16x16 loadA_f32row(const float* row, int hi, int kvalid) {
  bf16x16 a;
#pragma unroll
  for (int i = 0; i < 16; ++i) {
    int k = (i < 8) ? (hi * 8 + i) : (16 + hi * 8 + (i - 8));
    float v = (k < kvalid) ? row[k] : 0.0f;
    a[i] = (__bf16)v;
  }
  return a;
}

// B fragment: 16 contiguous bf16 per lane per tile (works for global or LDS ptr).
DEV_INLINE bf16x16 loadB_packed(const __bf16* packed, int tile, int lane) {
  union { bf16x16 v; uint4 q[2]; } u;
  const uint4* p = (const uint4*)(packed + ((size_t)tile * 32 + lane) * 16);
  u.q[0] = p[0];
  u.q[1] = p[1];
  return u.v;
}

DEV_INLINE unsigned int pack2bf16(float a, float b) {
  union { __bf16 h; unsigned short s; } lo, hi;
  lo.h = (__bf16)a; hi.h = (__bf16)b;
  return (unsigned int)lo.s | ((unsigned int)hi.s << 16);
}

// ---------------------------------------------------------------------------
// TDM: DMA nelem bf16 of packed weights (1-D tile) from global into LDS.
// D# built per CDNA5 ISA ch.8 (group0: count/type/addresses, group1: dims).
// ---------------------------------------------------------------------------
#if HAVE_TDM
DEV_INLINE void tdm_load_weights(const __bf16* gsrc, void* sdst, int nelem) {
  unsigned long long ga = (unsigned long long)(uintptr_t)gsrc;
  unsigned int lds_off = (unsigned int)(uintptr_t)sdst;  // low 32 bits = LDS offset
  union { u32x4 v; unsigned int w[4]; } g0;
  g0.w[0] = 1u;                                   // count=1, user descriptor
  g0.w[1] = lds_off;                              // lds_addr
  g0.w[2] = (unsigned int)(ga & 0xffffffffu);     // global_addr[95:64] -> addr[31:0]
  g0.w[3] = (unsigned int)((ga >> 32) & 0x01ffffffu) | 0x80000000u; // addr[56:32] | type=2
  unsigned long long ne = (unsigned long long)nelem;
  union { i32x8 v; unsigned long long q[4]; } g1;
  // [17:16] data_size=1 (2B); [79:48] tensor_dim0; [111:80] tensor_dim1=1;
  // [127:112] tile_dim0; [143:128] tile_dim1=1; [207:160] tensor_dim0_stride.
  g1.q[0] = (1ull << 16) | ((ne & 0xffffull) << 48);
  g1.q[1] = ((ne >> 16) & 0xffffull) | (1ull << 16) | ((ne & 0xffffull) << 48);
  g1.q[2] = 1ull | ((ne & 0xffffffffull) << 32);
  g1.q[3] = 0ull;
  i32x4 z4 = {0, 0, 0, 0};
#if __clang_major__ >= 23
  i32x8 z8 = {0, 0, 0, 0, 0, 0, 0, 0};
  __builtin_amdgcn_tensor_load_to_lds(g0.v, g1.v, z4, z4, z8, 0);
#else
  __builtin_amdgcn_tensor_load_to_lds(g0.v, g1.v, z4, z4, 0);
#endif
}
#endif

// Stage packed weights into LDS (TDM if available, cooperative copy otherwise).
// Caller must __syncthreads() afterwards.
DEV_INLINE void stage_weights(const __bf16* g, __bf16* s, int nelem) {
#if HAVE_TDM
  if (threadIdx.x < 32) {           // wave 0 issues the DMA
    tdm_load_weights(g, s, nelem);
#if HAVE_TENSORCNT
    __builtin_amdgcn_s_wait_tensorcnt(0);
#endif
  }
#else
  const unsigned int* gs = (const unsigned int*)g;
  unsigned int* ss = (unsigned int*)s;
  for (int i = threadIdx.x; i < (nelem >> 1); i += blockDim.x) ss[i] = gs[i];
#endif
}

// Pack row-major fp32 W[K x N] into WMMA B lane layout (bf16), zero-padded.
__global__ void pack_B_kernel(const float* __restrict__ W, __bf16* __restrict__ dst,
                              int K, int N, int KC, int NT) {
  int idx = blockIdx.x * blockDim.x + threadIdx.x;
  int total = KC * NT * 32 * 16;
  if (idx >= total) return;
  int i    = idx & 15;
  int lane = (idx >> 4) & 31;
  int tile = idx >> 9;
  int nt = tile % NT, kc = tile / NT;
  int k = kc * 32 + ((i < 8) ? ((lane >> 4) * 8 + i) : (16 + (lane >> 4) * 8 + (i - 8)));
  int n = nt * 16 + (lane & 15);
  float v = (k < K && n < N) ? W[(size_t)k * N + n] : 0.0f;
  dst[idx] = (__bf16)v;
}

// ---------------------------------------------------------------------------
// h = x[N x 32] @ node_W[32 x 64] + b      (store fp32 + bf16, coalesced)
// ---------------------------------------------------------------------------
__global__ __launch_bounds__(256)
void node_embed_kernel(const float* __restrict__ x, const __bf16* __restrict__ Wp,
                       const float* __restrict__ bias,
                       float* __restrict__ h_f32, __bf16* __restrict__ h_bf, int N) {
  __shared__ float sOut[8][1024];
  int wave = threadIdx.x >> 5, lane = threadIdx.x & 31;
  int tiles = N >> 4;                    // N % 16 == 0
  int tile = blockIdx.x * 8 + wave;
  bool wvalid = tile < tiles;
  int tc = wvalid ? tile : tiles - 1;
  int hi = lane >> 4, mrow = lane & 15;
  int node = tc * 16 + mrow;
  bf16x16 a = loadA_f32row(x + (size_t)node * 32, hi, 32);
  f32x8 acc[4] = {};
#pragma unroll
  for (int nt = 0; nt < 4; ++nt)
    acc[nt] = wmma_bf16(a, loadB_packed(Wp, nt, lane), acc[nt]);
  float* sw = &sOut[wave][0];
  int ncol = lane & 15;
#pragma unroll
  for (int nt = 0; nt < 4; ++nt) {
    int n = nt * 16 + ncol;
    float bb = bias[n];
#pragma unroll
    for (int r = 0; r < 8; ++r) sw[(r + hi * 8) * 64 + n] = acc[nt][r] + bb;
  }
  // coalesced writeback: lane covers row=lane/2, 32 cols
  int row2 = lane >> 1, cb = (lane & 1) * 32;
  if (wvalid) {
    const float* sp = sw + row2 * 64 + cb;
    size_t gbase = (size_t)(tc * 16 + row2) * 64 + cb;
    const float4* sp4 = (const float4*)sp;
    float4* hp = (float4*)(h_f32 + gbase);
#pragma unroll
    for (int j = 0; j < 8; ++j) hp[j] = sp4[j];
    union { uint4 q[4]; unsigned int w[16]; } ob;
#pragma unroll
    for (int j = 0; j < 16; ++j) ob.w[j] = pack2bf16(sp[2 * j], sp[2 * j + 1]);
    uint4* bp = (uint4*)(h_bf + gbase);
#pragma unroll
    for (int j = 0; j < 4; ++j) bp[j] = ob.q[j];
  }
}

// ---------------------------------------------------------------------------
// e = edge_attr[E x 16] @ edge_W[16 x 32] + b   (K padded to 32, bf16 out)
// ---------------------------------------------------------------------------
__global__ __launch_bounds__(256)
void edge_embed_kernel(const float* __restrict__ ea, const __bf16* __restrict__ Wp,
                       const float* __restrict__ bias, __bf16* __restrict__ e_bf, int E) {
  __shared__ float sOut[8][512];
  int wave = threadIdx.x >> 5, lane = threadIdx.x & 31;
  int tiles = E >> 4;                    // E % 16 == 0
  int tile = blockIdx.x * 8 + wave;
  bool wvalid = tile < tiles;
  int tc = wvalid ? tile : tiles - 1;
  int hi = lane >> 4, mrow = lane & 15;
  int edge = tc * 16 + mrow;
  bf16x16 a = loadA_f32row(ea + (size_t)edge * 16, hi, 16);
  f32x8 acc[2] = {};
#pragma unroll
  for (int nt = 0; nt < 2; ++nt)
    acc[nt] = wmma_bf16(a, loadB_packed(Wp, nt, lane), acc[nt]);
  float* sw = &sOut[wave][0];
  int ncol = lane & 15;
#pragma unroll
  for (int nt = 0; nt < 2; ++nt) {
    int n = nt * 16 + ncol;
    float bb = bias[n];
#pragma unroll
    for (int r = 0; r < 8; ++r) sw[(r + hi * 8) * 32 + n] = acc[nt][r] + bb;
  }
  int row2 = lane >> 1, cb = (lane & 1) * 16;
  if (wvalid) {
    const float* sp = sw + row2 * 32 + cb;
    union { uint4 q[2]; unsigned int w[8]; } ob;
#pragma unroll
    for (int j = 0; j < 8; ++j) ob.w[j] = pack2bf16(sp[2 * j], sp[2 * j + 1]);
    uint4* bp = (uint4*)(e_bf + (size_t)(tc * 16 + row2) * 32 + cb);
    bp[0] = ob.q[0];
    bp[1] = ob.q[1];
  }
}

// ---------------------------------------------------------------------------
// m = relu([h[src] | e] @ msg_W[96 x 64] + b); bf16 out; BN stats accumulated
// ---------------------------------------------------------------------------
__global__ __launch_bounds__(256)
void edge_msg_kernel(const __bf16* __restrict__ h_bf, const __bf16* __restrict__ e_bf,
                     const __bf16* __restrict__ Wp, const float* __restrict__ bias,
                     const int* __restrict__ srcI, __bf16* __restrict__ m_out,
                     float* __restrict__ gsum, float* __restrict__ gsq, int E) {
  __shared__ float sOut[8][1024];
  __shared__ __bf16 sB[12 * 512];
  __shared__ float s_sum[64], s_sq[64];
  stage_weights(Wp, sB, 12 * 512);
  if (threadIdx.x < 64) { s_sum[threadIdx.x] = 0.f; s_sq[threadIdx.x] = 0.f; }
  __syncthreads();

  int wave = threadIdx.x >> 5, lane = threadIdx.x & 31;
  int tiles = E >> 4;
  int tile = blockIdx.x * 8 + wave;
  bool wvalid = tile < tiles;
  int tc = wvalid ? tile : tiles - 1;
  int hi = lane >> 4, mrow = lane & 15;
  int edge = tc * 16 + mrow;
  int s = srcI[edge];
  const __bf16* hrow = h_bf + (size_t)s * 64;
  const __bf16* erow = e_bf + (size_t)edge * 32;
  f32x8 acc[4] = {};
#pragma unroll
  for (int kc = 0; kc < 3; ++kc) {
    bf16x16 a = (kc < 2) ? loadA_bf16row(hrow + kc * 32, hi)
                         : loadA_bf16row(erow, hi);
#pragma unroll
    for (int nt = 0; nt < 4; ++nt)
      acc[nt] = wmma_bf16(a, loadB_packed(sB, kc * 4 + nt, lane), acc[nt]);
  }
  float* sw = &sOut[wave][0];
  int ncol = lane & 15;
#pragma unroll
  for (int nt = 0; nt < 4; ++nt) {
    int n = nt * 16 + ncol;
    float bb = bias[n];
    float ps = 0.f, pq = 0.f;
#pragma unroll
    for (int r = 0; r < 8; ++r) {
      float v = acc[nt][r] + bb;
      v = v > 0.f ? v : 0.f;
      sw[(r + hi * 8) * 64 + n] = v;
      ps += v; pq += v * v;
    }
    if (wvalid) { atomicAdd(&s_sum[n], ps); atomicAdd(&s_sq[n], pq); }
  }
  int row2 = lane >> 1, cb = (lane & 1) * 32;
  if (wvalid) {
    const float* sp = sw + row2 * 64 + cb;
    union { uint4 q[4]; unsigned int w[16]; } ob;
#pragma unroll
    for (int j = 0; j < 16; ++j) ob.w[j] = pack2bf16(sp[2 * j], sp[2 * j + 1]);
    uint4* mp = (uint4*)(m_out + (size_t)(tc * 16 + row2) * 64 + cb);
#pragma unroll
    for (int j = 0; j < 4; ++j) mp[j] = ob.q[j];
  }
  __syncthreads();
  if (threadIdx.x < 64) {
    atomicAdd(&gsum[threadIdx.x], s_sum[threadIdx.x]);
    atomicAdd(&gsq[threadIdx.x], s_sq[threadIdx.x]);
  }
}

// scale = gamma*rsqrt(var+eps); shift = beta - mu*scale
__global__ void bn_finalize_kernel(const float* __restrict__ gsum, const float* __restrict__ gsq,
                                   const float* __restrict__ gamma, const float* __restrict__ beta,
                                   float inv_n, float* __restrict__ scale, float* __restrict__ shift) {
  int f = threadIdx.x;
  if (f >= 64) return;
  float mu = gsum[f] * inv_n;
  float var = gsq[f] * inv_n - mu * mu;
  float rs = rsqrtf(var + 1e-5f);
  float sc = gamma[f] * rs;
  scale[f] = sc;
  shift[f] = beta[f] - mu * sc;
}

// agg[dst] += BN(m) — affine folded into the scatter; 4 features per thread
__global__ void scatter_kernel(const __bf16* __restrict__ m, const int* __restrict__ dstI,
                               const float* __restrict__ scale, const float* __restrict__ shift,
                               float* __restrict__ agg, int E) {
  int idx = blockIdx.x * blockDim.x + threadIdx.x;
  if (idx >= E * 16) return;
  int e = idx >> 4, f = (idx & 15) * 4;
  __builtin_prefetch(m + (size_t)(e + 512) * 64 + f, 0, 1);
  union { uint2 u; __bf16 b[4]; } mv;
  mv.u = *(const uint2*)(m + (size_t)e * 64 + f);
  float* ap = agg + (size_t)dstI[e] * 64 + f;
#pragma unroll
  for (int j = 0; j < 4; ++j)
    atomicAdd(ap + j, (float)mv.b[j] * scale[f + j] + shift[f + j]);
}

// ---------------------------------------------------------------------------
// u = relu([h | agg] @ upd_W[128 x 64] + b); fp32 out; BN stats accumulated
// ---------------------------------------------------------------------------
__global__ __launch_bounds__(256)
void node_update_kernel(const __bf16* __restrict__ h_bf, const float* __restrict__ agg,
                        const __bf16* __restrict__ Wp, const float* __restrict__ bias,
                        float* __restrict__ u_out,
                        float* __restrict__ gsum, float* __restrict__ gsq, int N) {
  __shared__ float sOut[8][1024];
  __shared__ __bf16 sB[16 * 512];
  __shared__ float s_sum[64], s_sq[64];
  stage_weights(Wp, sB, 16 * 512);
  if (threadIdx.x < 64) { s_sum[threadIdx.x] = 0.f; s_sq[threadIdx.x] = 0.f; }
  __syncthreads();

  int wave = threadIdx.x >> 5, lane = threadIdx.x & 31;
  int tiles = N >> 4;
  int tile = blockIdx.x * 8 + wave;
  bool wvalid = tile < tiles;
  int tc = wvalid ? tile : tiles - 1;
  int hi = lane >> 4, mrow = lane & 15;
  int node = tc * 16 + mrow;
  const __bf16* hrow = h_bf + (size_t)node * 64;
  const float* arow = agg + (size_t)node * 64;
  f32x8 acc[4] = {};
#pragma unroll
  for (int kc = 0; kc < 4; ++kc) {
    bf16x16 a = (kc < 2) ? loadA_bf16row(hrow + kc * 32, hi)
                         : loadA_f32row(arow + (kc - 2) * 32, hi, 32);
#pragma unroll
    for (int nt = 0; nt < 4; ++nt)
      acc[nt] = wmma_bf16(a, loadB_packed(sB, kc * 4 + nt, lane), acc[nt]);
  }
  float* sw = &sOut[wave][0];
  int ncol = lane & 15;
#pragma unroll
  for (int nt = 0; nt < 4; ++nt) {
    int n = nt * 16 + ncol;
    float bb = bias[n];
    float ps = 0.f, pq = 0.f;
#pragma unroll
    for (int r = 0; r < 8; ++r) {
      float v = acc[nt][r] + bb;
      v = v > 0.f ? v : 0.f;
      sw[(r + hi * 8) * 64 + n] = v;
      ps += v; pq += v * v;
    }
    if (wvalid) { atomicAdd(&s_sum[n], ps); atomicAdd(&s_sq[n], pq); }
  }
  int row2 = lane >> 1, cb = (lane & 1) * 32;
  if (wvalid) {
    const float4* sp4 = (const float4*)(sw + row2 * 64 + cb);
    float4* up = (float4*)(u_out + (size_t)(tc * 16 + row2) * 64 + cb);
#pragma unroll
    for (int j = 0; j < 8; ++j) up[j] = sp4[j];
  }
  __syncthreads();
  if (threadIdx.x < 64) {
    atomicAdd(&gsum[threadIdx.x], s_sum[threadIdx.x]);
    atomicAdd(&gsq[threadIdx.x], s_sq[threadIdx.x]);
  }
}

// h = BN(u): apply affine, write fp32 + bf16 (4 features per thread)
__global__ void bn_apply_kernel(const float* __restrict__ u, const float* __restrict__ scale,
                                const float* __restrict__ shift,
                                float* __restrict__ h_f32, __bf16* __restrict__ h_bf, int N) {
  int idx = blockIdx.x * blockDim.x + threadIdx.x;
  if (idx >= N * 16) return;
  int f = (idx & 15) * 4;
  size_t base = (size_t)(idx >> 4) * 64 + f;
  float4 uv = *(const float4*)(u + base);
  float4 hv;
  hv.x = uv.x * scale[f + 0] + shift[f + 0];
  hv.y = uv.y * scale[f + 1] + shift[f + 1];
  hv.z = uv.z * scale[f + 2] + shift[f + 2];
  hv.w = uv.w * scale[f + 3] + shift[f + 3];
  *(float4*)(h_f32 + base) = hv;
  uint2 bv;
  bv.x = pack2bf16(hv.x, hv.y);
  bv.y = pack2bf16(hv.z, hv.w);
  *(uint2*)(h_bf + base) = bv;
}

// g[batch[n]] += h[n]   (4 features per thread)
__global__ void pool_kernel(const float* __restrict__ h, const int* __restrict__ batch,
                            float* __restrict__ g, int N) {
  int idx = blockIdx.x * blockDim.x + threadIdx.x;
  if (idx >= N * 16) return;
  int n = idx >> 4, f = (idx & 15) * 4;
  float4 hv = *(const float4*)(h + (size_t)n * 64 + f);
  float* gp = g + (size_t)batch[n] * 64 + f;
  atomicAdd(gp + 0, hv.x);
  atomicAdd(gp + 1, hv.y);
  atomicAdd(gp + 2, hv.z);
  atomicAdd(gp + 3, hv.w);
}

// out = relu(g @ r1_W + r1_b) @ r2_W + r2_b   (one block of 64 per graph)
__global__ void readout_kernel(const float* __restrict__ g, const float* __restrict__ r1W,
                               const float* __restrict__ r1b, const float* __restrict__ r2W,
                               const float* __restrict__ r2b, float* __restrict__ out) {
  __shared__ float sg[64], red[64];
  int f = threadIdx.x, gr = blockIdx.x;
  sg[f] = g[(size_t)gr * 64 + f];
  __syncthreads();
  float acc = r1b[f];
#pragma unroll 8
  for (int k = 0; k < 64; ++k) acc += sg[k] * r1W[k * 64 + f];
  acc = acc > 0.f ? acc : 0.f;
  red[f] = acc * r2W[f];
  __syncthreads();
  for (int s = 32; s > 0; s >>= 1) {
    if (f < s) red[f] += red[f + s];
    __syncthreads();
  }
  if (f == 0) out[gr] = red[0] + r2b[0];
}

// ---------------------------------------------------------------------------
extern "C" void kernel_launch(void* const* d_in, const int* in_sizes, int n_in,
                              void* d_out, int out_size, void* d_ws, size_t ws_size,
                              hipStream_t stream) {
  const float* x         = (const float*)d_in[0];
  const float* edge_attr = (const float*)d_in[1];
  const int*   edge_index= (const int*)d_in[2];
  const int*   batch     = (const int*)d_in[3];
  const float* node_W    = (const float*)d_in[4];
  const float* node_b    = (const float*)d_in[5];
  const float* edge_W    = (const float*)d_in[6];
  const float* edge_b    = (const float*)d_in[7];
  const float* msg_W     = (const float*)d_in[8];
  const float* msg_b     = (const float*)d_in[9];
  const float* msg_gamma = (const float*)d_in[10];
  const float* msg_beta  = (const float*)d_in[11];
  const float* upd_W     = (const float*)d_in[12];
  const float* upd_b     = (const float*)d_in[13];
  const float* upd_gamma = (const float*)d_in[14];
  const float* upd_beta  = (const float*)d_in[15];
  const float* r1_W = (const float*)d_in[16];
  const float* r1_b = (const float*)d_in[17];
  const float* r2_W = (const float*)d_in[18];
  const float* r2_b = (const float*)d_in[19];
  (void)in_sizes; (void)n_in; (void)out_size; (void)ws_size;

  const int N = 100000, E = 1600000, G = 1000;
  const int* srcI = edge_index;
  const int* dstI = edge_index + E;

  char* ws = (char*)d_ws;
  size_t off = 0;
  auto take = [&](size_t bytes) -> char* {
    char* p = ws + off;
    off += (bytes + 255) & ~(size_t)255;
    return p;
  };
  float*  h_f32  = (float*)take((size_t)N * 64 * 4);
  float*  u_f32  = (float*)take((size_t)N * 64 * 4);
  float*  agg    = (float*)take((size_t)N * 64 * 4);
  __bf16* h_bf   = (__bf16*)take((size_t)N * 64 * 2);
  __bf16* e_bf   = (__bf16*)take((size_t)E * 32 * 2);
  __bf16* m_bf   = (__bf16*)take((size_t)E * 64 * 2);
  float*  g_buf  = (float*)take((size_t)G * 64 * 4);
  float*  stats  = (float*)take(128 * 4);
  float*  bn_sc  = (float*)take(128 * 4);
  __bf16* nodeWp = (__bf16*)take((size_t)4 * 512 * 2);
  __bf16* edgeWp = (__bf16*)take((size_t)2 * 512 * 2);
  __bf16* msgWp  = (__bf16*)take((size_t)3 * 12 * 512 * 2);
  __bf16* updWp  = (__bf16*)take((size_t)3 * 16 * 512 * 2);

  float* gsum = stats;  float* gsq = stats + 64;
  float* scl  = bn_sc;  float* shf = bn_sc + 64;

  // --- pack weights into WMMA B layout (bf16) ---
  pack_B_kernel<<<(4 * 512 + 255) / 256, 256, 0, stream>>>(node_W, nodeWp, 32, 64, 1, 4);
  pack_B_kernel<<<(2 * 512 + 255) / 256, 256, 0, stream>>>(edge_W, edgeWp, 16, 32, 1, 2);
  for (int l = 0; l < 3; ++l) {
    pack_B_kernel<<<(12 * 512 + 255) / 256, 256, 0, stream>>>(
        msg_W + (size_t)l * 96 * 64, msgWp + (size_t)l * 12 * 512, 96, 64, 3, 4);
    pack_B_kernel<<<(16 * 512 + 255) / 256, 256, 0, stream>>>(
        upd_W + (size_t)l * 128 * 64, updWp + (size_t)l * 16 * 512, 128, 64, 4, 4);
  }

  // --- embeddings ---
  int ntiles = N / 16, etiles = E / 16;
  node_embed_kernel<<<(ntiles + 7) / 8, 256, 0, stream>>>(x, nodeWp, node_b, h_f32, h_bf, N);
  edge_embed_kernel<<<(etiles + 7) / 8, 256, 0, stream>>>(edge_attr, edgeWp, edge_b, e_bf, E);

  // --- message-passing layers ---
  for (int l = 0; l < 3; ++l) {
    hipMemsetAsync(stats, 0, 128 * 4, stream);
    edge_msg_kernel<<<(etiles + 7) / 8, 256, 0, stream>>>(
        h_bf, e_bf, msgWp + (size_t)l * 12 * 512, msg_b + l * 64, srcI, m_bf, gsum, gsq, E);
    bn_finalize_kernel<<<1, 64, 0, stream>>>(gsum, gsq, msg_gamma + l * 64, msg_beta + l * 64,
                                             1.0f / (float)E, scl, shf);
    hipMemsetAsync(agg, 0, (size_t)N * 64 * 4, stream);
    scatter_kernel<<<(E * 16 + 255) / 256, 256, 0, stream>>>(m_bf, dstI, scl, shf, agg, E);

    hipMemsetAsync(stats, 0, 128 * 4, stream);
    node_update_kernel<<<(ntiles + 7) / 8, 256, 0, stream>>>(
        h_bf, agg, updWp + (size_t)l * 16 * 512, upd_b + l * 64, u_f32, gsum, gsq, N);
    bn_finalize_kernel<<<1, 64, 0, stream>>>(gsum, gsq, upd_gamma + l * 64, upd_beta + l * 64,
                                             1.0f / (float)N, scl, shf);
    bn_apply_kernel<<<(N * 16 + 255) / 256, 256, 0, stream>>>(u_f32, scl, shf, h_f32, h_bf, N);
  }

  // --- readout ---
  hipMemsetAsync(g_buf, 0, (size_t)G * 64 * 4, stream);
  pool_kernel<<<(N * 16 + 255) / 256, 256, 0, stream>>>(h_f32, batch, g_buf, N);
  readout_kernel<<<G, 64, 0, stream>>>(g_buf, r1_W, r1_b, r2_W, r2_b, (float*)d_out);
}
